// HybridGVP_GAT_Long_Layer_67095979098197
// MI455X (gfx1250) — matrix operand order repaired and checked by reading, not compiled, not run on hardware
//
#include <hip/hip_runtime.h>

// ---------------- problem constants ----------------
constexpr int B  = 4;
constexpr int L  = 2048;
constexpr int D  = 256;
constexpr int H  = 4;
constexpr int R  = 128;
constexpr int E  = 32768;
constexpr int HD = D / H;     // 64
constexpr int FF = 4 * D;     // 1024
constexpr int ET = E + L;     // edges + self loops
constexpr int HDprod = H * D; // 1024
constexpr float NEG_SLOPE = 0.2f;

typedef __attribute__((ext_vector_type(16))) __bf16 v16bf;
typedef __attribute__((ext_vector_type(8)))  __bf16 bf16x8;
typedef __attribute__((ext_vector_type(8)))  float  v8f;

// ---------------- generic bf16 WMMA GEMM: C = act(A @ Wt^T + bias) ----------
// A: f32 row-major [M,K] (optionally + abias per K column)
// Wt: bf16 row-major [N,K]  (pre-transposed weight / activation)
// Requires M%64==0, N%64==0, K%32==0 (true for every GEMM in this layer).
constexpr int BM = 64, BN = 64, BK = 32;

template <int ACT, int ABIAS>
__global__ __launch_bounds__(256) void gemm_bf16_wmma(
    const float* __restrict__ A, const __bf16* __restrict__ Wt,
    const float* __restrict__ bias, const float* __restrict__ abias,
    float* __restrict__ C, int M, int N, int K, int lda) {
  __shared__ alignas(16) __bf16 As[BM * BK];  // [m][k]
  __shared__ alignas(16) __bf16 Bs[BN * BK];  // [n][k]
  const int tid  = threadIdx.x;
  const int lane = tid & 31;
  const int wave = tid >> 5;   // 0..7
  const int wm   = wave >> 1;  // 0..3 -> 16-row slice
  const int wn   = wave & 1;   // 0..1 -> 32-col slice (two 16-col fragments)
  const int m0   = blockIdx.y * BM;
  const int n0   = blockIdx.x * BN;

  // staging coordinates: thread -> 8 contiguous K elements of one row
  const int srow = tid >> 2;        // 0..63
  const int skk  = (tid & 3) * 8;   // 0,8,16,24
  const float*  Arow = A  + (size_t)(m0 + srow) * lda + skk;
  const __bf16* Wrow = Wt + (size_t)(n0 + srow) * K + skk;

  v8f acc0 = {0.f, 0.f, 0.f, 0.f, 0.f, 0.f, 0.f, 0.f};
  v8f acc1 = {0.f, 0.f, 0.f, 0.f, 0.f, 0.f, 0.f, 0.f};

  // per-lane fragment chunk indices (8-element = 16-byte units)
  const int am   = wm * 16 + (lane & 15);
  const int ksel = (lane >> 4) & 1;        // lane half selects K group
  const int ac0  = am * 4 + 2 * ksel;      // A chunks {2ksel, 2ksel+1}
  const int nr0  = wn * 32 + (lane & 15);  // B fragment rows
  const int nr1  = nr0 + 16;
  const int bc0  = nr0 * 4 + 2 * ksel;     // B chunks {2ksel, 2ksel+1}
  const int bc1  = nr1 * 4 + 2 * ksel;

  bf16x8* AsV = (bf16x8*)As;
  bf16x8* BsV = (bf16x8*)Bs;

  for (int k0 = 0; k0 < K; k0 += BK) {
    // ---- stage A: two b128 loads + pack-convert + one b128 LDS store ----
    float4 a01 = *(const float4*)(Arow + k0);
    float4 a23 = *(const float4*)(Arow + k0 + 4);
    if (ABIAS) {
      float4 b01 = *(const float4*)(abias + k0 + skk);
      float4 b23 = *(const float4*)(abias + k0 + skk + 4);
      a01.x += b01.x; a01.y += b01.y; a01.z += b01.z; a01.w += b01.w;
      a23.x += b23.x; a23.y += b23.y; a23.z += b23.z; a23.w += b23.w;
    }
    bf16x8 av;
    av[0] = (__bf16)a01.x; av[1] = (__bf16)a01.y;
    av[2] = (__bf16)a01.z; av[3] = (__bf16)a01.w;
    av[4] = (__bf16)a23.x; av[5] = (__bf16)a23.y;
    av[6] = (__bf16)a23.z; av[7] = (__bf16)a23.w;
    AsV[tid] = av;
    // ---- stage B: one b128 load + one b128 LDS store (already [n][k]) ----
    BsV[tid] = *(const bf16x8*)(Wrow + k0);
    __syncthreads();

    // ---- fragments: 6x ds_load_b128, 2x wmma ----
    bf16x8 a_lo = ((const bf16x8*)As)[ac0];
    bf16x8 a_hi = ((const bf16x8*)As)[ac0 + 1];
    bf16x8 b0l  = ((const bf16x8*)Bs)[bc0];
    bf16x8 b0h  = ((const bf16x8*)Bs)[bc0 + 1];
    bf16x8 b1l  = ((const bf16x8*)Bs)[bc1];
    bf16x8 b1h  = ((const bf16x8*)Bs)[bc1 + 1];
    v16bf af = __builtin_shufflevector(a_lo, a_hi, 0, 1, 2, 3, 4, 5, 6, 7, 8,
                                       9, 10, 11, 12, 13, 14, 15);
    v16bf bf0 = __builtin_shufflevector(b0l, b0h, 0, 1, 2, 3, 4, 5, 6, 7, 8,
                                        9, 10, 11, 12, 13, 14, 15);
    v16bf bf1 = __builtin_shufflevector(b1l, b1h, 0, 1, 2, 3, 4, 5, 6, 7, 8,
                                        9, 10, 11, 12, 13, 14, 15);
    acc0 = __builtin_amdgcn_wmma_f32_16x16x32_bf16(false, af, false, bf0,
                                                   (short)0, acc0, false, false);
    acc1 = __builtin_amdgcn_wmma_f32_16x16x32_bf16(false, af, false, bf1,
                                                   (short)0, acc1, false, false);
    __syncthreads();
  }

  // C/D layout: vgpr r -> M = r (lanes 0-15) or 8+r (lanes 16-31), N = lane&15
  const int ng0 = n0 + nr0;
  const int ng1 = n0 + nr1;
  const float bv0 = bias ? bias[ng0] : 0.f;
  const float bv1 = bias ? bias[ng1] : 0.f;
#pragma unroll
  for (int r = 0; r < 8; ++r) {
    int mrow = m0 + wm * 16 + ((lane & 16) ? (8 + r) : r);
    float v0 = acc0[r] + bv0;
    float v1 = acc1[r] + bv1;
    if (ACT == 1) { v0 = fmaxf(v0, 0.f); v1 = fmaxf(v1, 0.f); }
    C[(size_t)mrow * N + ng0] = v0;
    C[(size_t)mrow * N + ng1] = v1;
  }
}

// ---------------- small helpers ----------------
// transpose + convert: in f32 [K,N] -> out bf16 [N,K]
__global__ void transpose_cvt_kernel(const float* __restrict__ in,
                                     __bf16* __restrict__ out, int K, int N) {
  int i = blockIdx.x * 256 + threadIdx.x;
  if (i >= K * N) return;
  int k = i / N, n = i % N;  // consecutive tid -> consecutive n (coalesced read)
  out[(size_t)n * K + k] = (__bf16)in[i];
}

__global__ void fill_f32_kernel(float* __restrict__ p, float v, int n) {
  int i = blockIdx.x * 256 + threadIdx.x;
  if (i < n) p[i] = v;
}

static __device__ __forceinline__ void atomicMaxF(float* addr, float val) {
  int* ia  = (int*)addr;
  int  old = __float_as_int(*addr);
  while (__int_as_float(old) < val) {
    int prev = atomicCAS(ia, old, __float_as_int(val));
    if (prev == old) break;
    old = prev;
  }
}

// ---------------- GAT pieces ----------------
__global__ __launch_bounds__(128) void gat_coef_kernel(
    const float* __restrict__ h, const float* __restrict__ att_src,
    const float* __restrict__ att_dst, float* __restrict__ asrc,
    float* __restrict__ adst) {
  const int node = blockIdx.x;  // b*L + l
  const int head = threadIdx.x >> 5;
  const int lane = threadIdx.x & 31;
  const float* hp = h + (size_t)node * HDprod + head * D;
  float s1 = 0.f, s2 = 0.f;
  for (int c = lane; c < D; c += 32) {
    float hv = hp[c];
    s1 += hv * att_src[head * D + c];
    s2 += hv * att_dst[head * D + c];
  }
  for (int off = 16; off; off >>= 1) {
    s1 += __shfl_xor(s1, off, 32);
    s2 += __shfl_xor(s2, off, 32);
  }
  if (lane == 0) {
    asrc[(size_t)node * H + head] = s1;
    adst[(size_t)node * H + head] = s2;
  }
}

__global__ void gat_edge_max_kernel(const int* __restrict__ ei,
                                    const float* __restrict__ asrc,
                                    const float* __restrict__ adst,
                                    float* __restrict__ evals,
                                    float* __restrict__ mbuf) {
  int idx = blockIdx.x * 256 + threadIdx.x;  // [B*ET*H]
  if (idx >= B * ET * H) return;
  int hh = idx % H, rest = idx / H;
  int e = rest % ET, b = rest / ET;
  int s = (e < E) ? ei[e] : (e - E);
  int d = (e < E) ? ei[E + e] : (e - E);
  float v = asrc[((size_t)b * L + s) * H + hh] +
            adst[((size_t)b * L + d) * H + hh];
  v = (v > 0.f) ? v : NEG_SLOPE * v;
  evals[idx] = v;
  atomicMaxF(&mbuf[((size_t)b * L + d) * H + hh], v);
}

__global__ void gat_edge_exp_kernel(const int* __restrict__ ei,
                                    float* __restrict__ evals,
                                    const float* __restrict__ mbuf,
                                    float* __restrict__ den) {
  int idx = blockIdx.x * 256 + threadIdx.x;
  if (idx >= B * ET * H) return;
  int hh = idx % H, rest = idx / H;
  int e = rest % ET, b = rest / ET;
  int d = (e < E) ? ei[E + e] : (e - E);
  float ex = __expf(evals[idx] - mbuf[((size_t)b * L + d) * H + hh]);
  evals[idx] = ex;
  atomicAdd(&den[((size_t)b * L + d) * H + hh], ex);
}

__global__ __launch_bounds__(128) void gat_scatter_kernel(
    const int* __restrict__ ei, const float* __restrict__ evals,
    const float* __restrict__ den, const float* __restrict__ h,
    float* __restrict__ gout) {
  const int be   = blockIdx.x;  // b*ET + e
  const int b    = be / ET;
  const int e    = be % ET;
  const int head = threadIdx.x >> 5;
  const int lane = threadIdx.x & 31;
  int s = (e < E) ? ei[e] : (e - E);
  int d = (e < E) ? ei[E + e] : (e - E);
  float alpha = evals[((size_t)be) * H + head] /
                den[((size_t)b * L + d) * H + head];
  const float* hp = h + ((size_t)b * L + s) * HDprod + head * D;
  float* op = gout + ((size_t)b * L + d) * HDprod + head * D;
  for (int c = lane; c < D; c += 32) atomicAdd(&op[c], hp[c] * alpha);
}

// ---------------- layernorm over last dim (D=256), up to 3 summed inputs ----
__global__ __launch_bounds__(256) void layernorm3_kernel(
    const float* __restrict__ x0, const float* __restrict__ x1,
    const float* __restrict__ x2, const float* __restrict__ g,
    const float* __restrict__ bta, float* __restrict__ out, int rows) {
  int row = blockIdx.x * 8 + (threadIdx.x >> 5);
  if (row >= rows) return;
  int lane = threadIdx.x & 31;
  const float* p0 = x0 + (size_t)row * D;
  const float* p1 = x1 ? x1 + (size_t)row * D : nullptr;
  const float* p2 = x2 ? x2 + (size_t)row * D : nullptr;
  float xs[8], s = 0.f, s2 = 0.f;
#pragma unroll
  for (int j = 0; j < 8; ++j) {
    int c = lane + 32 * j;
    float v = p0[c];
    if (p1) v += p1[c];
    if (p2) v += p2[c];
    xs[j] = v; s += v; s2 += v * v;
  }
  for (int off = 16; off; off >>= 1) {
    s += __shfl_xor(s, off, 32);
    s2 += __shfl_xor(s2, off, 32);
  }
  float mu = s / (float)D;
  float var = s2 / (float)D - mu * mu;
  float rs = rsqrtf(var + 1e-5f);
#pragma unroll
  for (int j = 0; j < 8; ++j) {
    int c = lane + 32 * j;
    out[(size_t)row * D + c] = (xs[j] - mu) * rs * g[c] + bta[c];
  }
}

// -------- softmax over sequence axis; writes result TRANSPOSED --------------
// in : pm  [B*L, R]   (GEMM output)
// out: pmT [B, R, L]  (softmax over L, contiguous in L for the next GEMM)
__global__ __launch_bounds__(256) void softmax_seq_kernel(
    const float* __restrict__ pm, float* __restrict__ pmT) {
  const int b = blockIdx.x / R;
  const int r = blockIdx.x % R;
  __shared__ float red[256];
  const int tid = threadIdx.x;
  float m = -3e38f;
  for (int l = tid; l < L; l += 256)
    m = fmaxf(m, pm[((size_t)b * L + l) * R + r]);
  red[tid] = m; __syncthreads();
  for (int s = 128; s; s >>= 1) {
    if (tid < s) red[tid] = fmaxf(red[tid], red[tid + s]);
    __syncthreads();
  }
  m = red[0]; __syncthreads();
  float sum = 0.f;
  for (int l = tid; l < L; l += 256)
    sum += __expf(pm[((size_t)b * L + l) * R + r] - m);
  red[tid] = sum; __syncthreads();
  for (int s = 128; s; s >>= 1) {
    if (tid < s) red[tid] += red[tid + s];
    __syncthreads();
  }
  float inv = 1.f / red[0];
  float* orow = pmT + ((size_t)b * R + r) * L;
  for (int l = tid; l < L; l += 256)
    orow[l] = __expf(pm[((size_t)b * L + l) * R + r] - m) * inv;
}

// ---------------- compressed attention: wave per (b,l,h), R=128, HD=64 ------
__global__ __launch_bounds__(256) void long_attn_kernel(
    const float* __restrict__ q, const float* __restrict__ kc,
    const float* __restrict__ vc, float* __restrict__ lng) {
  __shared__ float qs[8][HD];
  __shared__ float ps[8][R];
  const int wave = threadIdx.x >> 5;
  const int lane = threadIdx.x & 31;
  const int gid  = blockIdx.x * 8 + wave;  // b*L*H + l*H + h
  const int hh   = gid % H;
  const int rest = gid / H;
  const int l    = rest % L;
  const int b    = rest / L;

  const float* qp = q + ((size_t)b * L + l) * D + hh * HD;
  qs[wave][lane]      = qp[lane];
  qs[wave][lane + 32] = qp[lane + 32];
  __syncthreads();

  float sc[4];
#pragma unroll
  for (int j = 0; j < 4; ++j) {
    int r = lane + 32 * j;
    const float* kp = kc + ((size_t)b * R + r) * D + hh * HD;
    float acc = 0.f;
    for (int d = 0; d < HD; ++d) acc += qs[wave][d] * kp[d];
    sc[j] = acc * 0.125f;  // 1/sqrt(64)
  }
  float m = fmaxf(fmaxf(sc[0], sc[1]), fmaxf(sc[2], sc[3]));
  for (int off = 16; off; off >>= 1) m = fmaxf(m, __shfl_xor(m, off, 32));
  float sum = 0.f;
#pragma unroll
  for (int j = 0; j < 4; ++j) { sc[j] = __expf(sc[j] - m); sum += sc[j]; }
  for (int off = 16; off; off >>= 1) sum += __shfl_xor(sum, off, 32);
  float inv = 1.f / sum;
#pragma unroll
  for (int j = 0; j < 4; ++j) ps[wave][lane + 32 * j] = sc[j] * inv;
  __syncthreads();

  float o0 = 0.f, o1 = 0.f;
  for (int r = 0; r < R; ++r) {
    float p = ps[wave][r];
    const float* vp = vc + ((size_t)b * R + r) * D + hh * HD;
    o0 += p * vp[lane];
    o1 += p * vp[lane + 32];
  }
  float* op = lng + ((size_t)b * L + l) * D + hh * HD;
  op[lane]      = o0;
  op[lane + 32] = o1;
}

// ---------------- host side ----------------
static inline size_t alignUp(size_t x) { return (x + 255) & ~(size_t)255; }

extern "C" void kernel_launch(void* const* d_in, const int* in_sizes, int n_in,
                              void* d_out, int out_size, void* d_ws,
                              size_t ws_size, hipStream_t stream) {
  (void)in_sizes; (void)n_in; (void)out_size; (void)ws_size;
  const float* s         = (const float*)d_in[0];
  const float* gat_W     = (const float*)d_in[1];
  const float* att_src   = (const float*)d_in[2];
  const float* att_dst   = (const float*)d_in[3];
  const float* gat_b     = (const float*)d_in[4];
  const float* short_W   = (const float*)d_in[5];
  const float* short_b   = (const float*)d_in[6];
  const float* q_W = (const float*)d_in[7];  const float* q_b = (const float*)d_in[8];
  const float* k_W = (const float*)d_in[9];  const float* k_b = (const float*)d_in[10];
  const float* v_W = (const float*)d_in[11]; const float* v_b = (const float*)d_in[12];
  const float* o_W = (const float*)d_in[13]; const float* o_b = (const float*)d_in[14];
  const float* p_W = (const float*)d_in[15]; const float* p_b = (const float*)d_in[16];
  const float* lnq_g = (const float*)d_in[17]; const float* lnq_b = (const float*)d_in[18];
  const float* lnk_g = (const float*)d_in[19]; const float* lnk_b = (const float*)d_in[20];
  const float* lnv_g = (const float*)d_in[21]; const float* lnv_b = (const float*)d_in[22];
  const float* ln1_g = (const float*)d_in[23]; const float* ln1_b = (const float*)d_in[24];
  const float* ln2_g = (const float*)d_in[25]; const float* ln2_b = (const float*)d_in[26];
  const float* f1_W = (const float*)d_in[27]; const float* f1_b = (const float*)d_in[28];
  const float* f2_W = (const float*)d_in[29]; const float* f2_b = (const float*)d_in[30];
  const int*   ei   = (const int*)d_in[31];   // [2, E]
  float* out = (float*)d_out;

  const int BL  = B * L;       // 8192 rows
  const int BLD = B * L * D;   // 2,097,152

  // ---- workspace layout ----
  char*  base = (char*)d_ws;
  size_t off  = 0;
  auto allocB = [&](size_t bytes) { void* p = base + off; off = alignUp(off + bytes); return p; };

  // all B-operands stored transposed: [N, K] bf16
  __bf16* gatWt   = (__bf16*)allocB(sizeof(__bf16) * HDprod * D);  // [1024,256]
  __bf16* shortWt = (__bf16*)allocB(sizeof(__bf16) * D * HDprod);  // [256,1024]
  __bf16* qWt     = (__bf16*)allocB(sizeof(__bf16) * D * D);
  __bf16* kWt     = (__bf16*)allocB(sizeof(__bf16) * D * D);
  __bf16* vWt     = (__bf16*)allocB(sizeof(__bf16) * D * D);
  __bf16* oWt     = (__bf16*)allocB(sizeof(__bf16) * D * D);
  __bf16* pWt     = (__bf16*)allocB(sizeof(__bf16) * R * D);       // [128,256]
  __bf16* f1Wt    = (__bf16*)allocB(sizeof(__bf16) * FF * D);      // [1024,256]
  __bf16* f2Wt    = (__bf16*)allocB(sizeof(__bf16) * D * FF);      // [256,1024]
  __bf16* kbfT    = (__bf16*)allocB(sizeof(__bf16) * BLD);         // [B][D][L]
  __bf16* vbfT    = (__bf16*)allocB(sizeof(__bf16) * BLD);         // [B][D][L]

  float* hbuf   = (float*)allocB(sizeof(float) * (size_t)BL * HDprod);  // reused as ff1
  float* gout   = (float*)allocB(sizeof(float) * (size_t)BL * HDprod);  // reused (see below)
  float* asrc   = (float*)allocB(sizeof(float) * BL * H);
  float* adst   = (float*)allocB(sizeof(float) * BL * H);
  float* mbuf   = (float*)allocB(sizeof(float) * BL * H);
  float* den    = (float*)allocB(sizeof(float) * BL * H);
  float* evals  = (float*)allocB(sizeof(float) * (size_t)B * ET * H);
  float* shortb = (float*)allocB(sizeof(float) * BLD);
  float* qb     = (float*)allocB(sizeof(float) * BLD);
  float* kb     = (float*)allocB(sizeof(float) * BLD);
  float* vb     = (float*)allocB(sizeof(float) * BLD);
  float* pm     = (float*)allocB(sizeof(float) * (size_t)BL * R);
  float* pmT    = (float*)allocB(sizeof(float) * (size_t)B * R * L);
  float* kc     = (float*)allocB(sizeof(float) * B * R * D);
  float* vc     = (float*)allocB(sizeof(float) * B * R * D);
  // alias region: gout (BL*1024 floats = 4*BLD) reused after short GEMM
  float* lng  = gout;
  float* lngo = gout + (size_t)BLD;
  float* h1   = gout + (size_t)2 * BLD;
  float* ff2  = gout + (size_t)3 * BLD;
  float* ff1  = hbuf;  // BL*FF, reused after GAT scatter

  auto tcvt = [&](const float* src_, __bf16* dst_, int K, int N) {
    transpose_cvt_kernel<<<(K * N + 255) / 256, 256, 0, stream>>>(src_, dst_, K, N);
  };
  auto gemm = [&](const float* A, const __bf16* Wt, const float* bias,
                  const float* abias, float* C, int M, int N, int K, int lda,
                  int act) {
    dim3 g(N / BN, M / BM);
    if (abias)
      gemm_bf16_wmma<0, 1><<<g, 256, 0, stream>>>(A, Wt, bias, abias, C, M, N, K, lda);
    else if (act)
      gemm_bf16_wmma<1, 0><<<g, 256, 0, stream>>>(A, Wt, bias, abias, C, M, N, K, lda);
    else
      gemm_bf16_wmma<0, 0><<<g, 256, 0, stream>>>(A, Wt, bias, abias, C, M, N, K, lda);
  };

  // ---- weight conversion (transpose to [N,K] bf16) ----
  tcvt(gat_W, gatWt, D, HDprod);
  tcvt(short_W, shortWt, HDprod, D);
  tcvt(q_W, qWt, D, D); tcvt(k_W, kWt, D, D); tcvt(v_W, vWt, D, D);
  tcvt(o_W, oWt, D, D); tcvt(p_W, pWt, D, R);
  tcvt(f1_W, f1Wt, D, FF); tcvt(f2_W, f2Wt, FF, D);

  // ---- GAT branch ----
  gemm(s, gatWt, nullptr, nullptr, hbuf, BL, HDprod, D, D, 0);  // h = s @ gat_W
  gat_coef_kernel<<<BL, 128, 0, stream>>>(hbuf, att_src, att_dst, asrc, adst);
  fill_f32_kernel<<<(BL * H + 255) / 256, 256, 0, stream>>>(mbuf, -3e38f, BL * H);
  fill_f32_kernel<<<(BL * H + 255) / 256, 256, 0, stream>>>(den, 0.f, BL * H);
  fill_f32_kernel<<<(BL * HDprod + 255) / 256, 256, 0, stream>>>(gout, 0.f, BL * HDprod);
  {
    int tot = B * ET * H;
    gat_edge_max_kernel<<<(tot + 255) / 256, 256, 0, stream>>>(ei, asrc, adst, evals, mbuf);
    gat_edge_exp_kernel<<<(tot + 255) / 256, 256, 0, stream>>>(ei, evals, mbuf, den);
  }
  gat_scatter_kernel<<<B * ET, 128, 0, stream>>>(ei, evals, den, hbuf, gout);
  // short = (gat_out + gat_b) @ short_W + short_b   (gat_b folded as A-column bias)
  gemm(gout, shortWt, short_b, gat_b, shortb, BL, D, HDprod, HDprod, 0);

  // ---- long-range branch ----
  gemm(s, qWt, q_b, nullptr, qb, BL, D, D, D, 0);
  gemm(s, kWt, k_b, nullptr, kb, BL, D, D, D, 0);
  gemm(s, vWt, v_b, nullptr, vb, BL, D, D, D, 0);
  layernorm3_kernel<<<BL / 8, 256, 0, stream>>>(qb, nullptr, nullptr, lnq_g, lnq_b, qb, BL);
  layernorm3_kernel<<<BL / 8, 256, 0, stream>>>(kb, nullptr, nullptr, lnk_g, lnk_b, kb, BL);
  layernorm3_kernel<<<BL / 8, 256, 0, stream>>>(vb, nullptr, nullptr, lnv_g, lnv_b, vb, BL);
  // transposed bf16 activations for the compression GEMMs: [D, L] per batch
  for (int b = 0; b < B; ++b) {
    tcvt(kb + (size_t)b * L * D, kbfT + (size_t)b * D * L, L, D);
    tcvt(vb + (size_t)b * L * D, vbfT + (size_t)b * D * L, L, D);
  }

  gemm(kb, pWt, p_b, nullptr, pm, BL, R, D, D, 0);           // [B*L, R]
  softmax_seq_kernel<<<B * R, 256, 0, stream>>>(pm, pmT);    // softmax over L -> [B,R,L]

  for (int b = 0; b < B; ++b) {  // k_c = pm^T @ k, v_c = pm^T @ v  (per batch)
    gemm(pmT + (size_t)b * R * L, kbfT + (size_t)b * D * L, nullptr, nullptr,
         kc + (size_t)b * R * D, R, D, L, L, 0);
    gemm(pmT + (size_t)b * R * L, vbfT + (size_t)b * D * L, nullptr, nullptr,
         vc + (size_t)b * R * D, R, D, L, L, 0);
  }

  long_attn_kernel<<<(B * L * H) / 8, 256, 0, stream>>>(qb, kc, vc, lng);
  gemm(lng, oWt, o_b, nullptr, lngo, BL, D, D, D, 0);

  // ---- fuse + FFN ----
  layernorm3_kernel<<<BL / 8, 256, 0, stream>>>(s, shortb, lngo, ln1_g, ln1_b, h1, BL);
  gemm(h1, f1Wt, f1_b, nullptr, ff1, BL, FF, D, D, 1);       // relu
  gemm(ff1, f2Wt, f2_b, nullptr, ff2, BL, D, FF, FF, 0);
  layernorm3_kernel<<<BL / 8, 256, 0, stream>>>(h1, ff2, nullptr, ln2_g, ln2_b, out, BL);
}
